// model2_2l_30073361006598
// MI455X (gfx1250) — matrix operand order repaired and checked by reading, not compiled, not run on hardware
//
#include <hip/hip_runtime.h>
#include <hip/hip_bf16.h>

typedef __attribute__((ext_vector_type(16))) _Float16 v16h;
typedef __attribute__((ext_vector_type(8)))  float    v8f;

#define N_GRAPHS 64
#define EPS_BN 1e-5f

// ---------------------------------------------------------------- utilities
__global__ void k_zero(float* __restrict__ p, int n) {
    int i = blockIdx.x * blockDim.x + threadIdx.x;
    int stride = gridDim.x * blockDim.x;
    for (; i < n; i += stride) p[i] = 0.0f;
}

// ---------------------------------------------------------------- layer 1 conv
// msg[e,o] = sum_{k,i} basis[e,k] * x[src[e],i] * W1[k,i,o]   (i<3, k<4, o<16)
// A (16 edges x 32) f16, column c = k*8 + i (zero-padded i>=3); B (32x16) from W1.
// One v_wmma_f32_16x16x32_f16 per 16-edge tile.
template<bool FULL>
__device__ __forceinline__ void conv1_tile(
    int tile, int nEdges, int m, int half,
    const float* __restrict__ x, const long long* __restrict__ ei,
    const float* __restrict__ ea, const v16h& bm,
    float* __restrict__ agg1, float* __restrict__ cnt)
{
    int  e  = tile * 16 + m;
    bool ev = FULL ? true : (e < nEdges);
    int  ec = FULL ? e : (ev ? e : (nEdges - 1));

    int src = (int)__builtin_nontemporal_load(ei + ec);
    int dst = (int)__builtin_nontemporal_load(ei + nEdges + ec);
    float p0 = __builtin_nontemporal_load(ea + 2 * ec + 0);
    float p1 = __builtin_nontemporal_load(ea + 2 * ec + 1);

    float b0 = (1.0f - p1) * (1.0f - p0);
    float b1 = (1.0f - p1) * p0;
    float b2 = p1 * (1.0f - p0);
    float b3 = p1 * p0;
    float ba = half ? b1 : b0;    // halves hold k={0,2} vs k={1,3} slices
    float bb = half ? b3 : b2;

    float x0 = x[src * 3 + 0];
    float x1 = x[src * 3 + 1];
    float x2 = x[src * 3 + 2];
    if (!FULL && !ev) { x0 = 0.0f; x1 = 0.0f; x2 = 0.0f; }

    v16h am;
    #pragma unroll
    for (int t = 0; t < 16; ++t) am[t] = (_Float16)0.0f;
    am[0] = (_Float16)(ba * x0); am[1] = (_Float16)(ba * x1); am[2]  = (_Float16)(ba * x2);
    am[8] = (_Float16)(bb * x0); am[9] = (_Float16)(bb * x1); am[10] = (_Float16)(bb * x2);

    v8f c;
    #pragma unroll
    for (int t = 0; t < 8; ++t) c[t] = 0.0f;
    c = __builtin_amdgcn_wmma_f32_16x16x32_f16(false, am, false, bm,
                                               (short)0, c, false, false);

    // C layout: lane holds column o=m, rows M = half*8 + j. Batch all row-dst
    // fetches (one DS wait), then straight-line scatter atomics.
    int dsts[8];
    #pragma unroll
    for (int j = 0; j < 8; ++j)
        dsts[j] = __builtin_amdgcn_ds_bpermute((half * 8 + j) << 2, dst);

    #pragma unroll
    for (int j = 0; j < 8; ++j) {
        if (FULL || (tile * 16 + half * 8 + j) < nEdges)
            atomicAdd(&agg1[dsts[j] * 16 + m], c[j]);
    }
    if (half == 0 && ev) atomicAdd(&cnt[dst], 1.0f);
}

__global__ void k_conv1(const float* __restrict__ x,
                        const long long* __restrict__ ei,   // [2, nEdges] int64
                        const float* __restrict__ ea,       // [nEdges, 2]
                        const float* __restrict__ W1,       // [4,3,16]
                        float* __restrict__ agg1,           // [nNodes,16]
                        float* __restrict__ cnt,            // [nNodes]
                        int nEdges)
{
    const int lane = threadIdx.x & 31;
    const int m    = lane & 15;       // A-row / B-col / C-col index
    const int half = lane >> 4;       // which half-wave

    // B in wave32 16-bit B layout: vgpr j holds K = half*8+2j (+8 for j>=4), +1 for col N=m.
    v16h bm;
    #pragma unroll
    for (int j = 0; j < 8; ++j) {
        int r0 = half * 8 + 2 * j + (j >= 4 ? 8 : 0);
        int r1 = r0 + 1;
        int k0 = r0 >> 3, i0 = r0 & 7;
        int k1 = r1 >> 3, i1 = r1 & 7;
        float v0 = (i0 < 3) ? W1[k0 * 48 + i0 * 16 + m] : 0.0f;
        float v1 = (i1 < 3) ? W1[k1 * 48 + i1 * 16 + m] : 0.0f;
        bm[2 * j]     = (_Float16)v0;
        bm[2 * j + 1] = (_Float16)v1;
    }

    const int wave   = (int)((blockIdx.x * blockDim.x + threadIdx.x) >> 5);
    const int nWaves = (int)((gridDim.x * blockDim.x) >> 5);
    const int nTiles = (nEdges + 15) >> 4;
    const int nFull  = nEdges >> 4;   // tiles with all 16 edges valid

    for (int tile = wave; tile < nFull; tile += nWaves)
        conv1_tile<true>(tile, nEdges, m, half, x, ei, ea, bm, agg1, cnt);
    for (int tile = nFull + wave; tile < nTiles; tile += nWaves)
        conv1_tile<false>(tile, nEdges, m, half, x, ei, ea, bm, agg1, cnt);
}

// ---------------------------------------------------------------- post layer1
// h1 = elu(agg1 / max(cnt,1)); accumulate per-feature sum / sumsq for BN.
__global__ void k_post1(const float* __restrict__ agg1,
                        const float* __restrict__ cnt,
                        float* __restrict__ h1,
                        float* __restrict__ stats1,   // [32] = sum[16], sumsq[16]
                        int nNodes)
{
    __shared__ float red[256];
    const int f    = threadIdx.x & 15;
    const int nl   = threadIdx.x >> 4;      // 0..15
    const int base = blockIdx.x * 32;

    float s = 0.0f, s2 = 0.0f;
    #pragma unroll
    for (int r = 0; r < 2; ++r) {
        int n = base + nl + 16 * r;
        if (n < nNodes) {
            float cv = fmaxf(cnt[n], 1.0f);
            float v  = agg1[n * 16 + f] / cv;
            float e  = v > 0.0f ? v : (expf(v) - 1.0f);
            h1[n * 16 + f] = e;
            s += e; s2 += e * e;
        }
    }
    red[threadIdx.x] = s; __syncthreads();
    for (int st = 128; st >= 16; st >>= 1) {
        if ((int)threadIdx.x < st) red[threadIdx.x] += red[threadIdx.x + st];
        __syncthreads();
    }
    if (threadIdx.x < 16) atomicAdd(&stats1[f], red[threadIdx.x]);
    __syncthreads();
    red[threadIdx.x] = s2; __syncthreads();
    for (int st = 128; st >= 16; st >>= 1) {
        if ((int)threadIdx.x < st) red[threadIdx.x] += red[threadIdx.x + st];
        __syncthreads();
    }
    if (threadIdx.x < 16) atomicAdd(&stats1[16 + f], red[threadIdx.x]);
}

// stats -> per-feature affine (scale = g*rsqrt(var+eps), shift = b - mu*scale)
__global__ void k_fin1(const float* __restrict__ stats1,
                       const float* __restrict__ g1, const float* __restrict__ b1,
                       float* __restrict__ ss1, int nNodes)
{
    int t = threadIdx.x;
    if (t < 16) {
        float invn = 1.0f / (float)nNodes;
        float mu  = stats1[t] * invn;
        float var = stats1[16 + t] * invn - mu * mu;
        float inv = rsqrtf(var + EPS_BN);
        float sc  = inv * g1[t];
        ss1[t]      = sc;
        ss1[16 + t] = b1[t] - mu * sc;
    }
}

// ---------------------------------------------------------------- layer 2 conv
// Input features = BN1(h1) applied on gather. A is 16x64 f16 (col = k*16+i),
// B = W2 reshaped 64x32. 4x v_wmma_f32_16x16x32_f16 per 16-edge tile.
template<bool FULL>
__device__ __forceinline__ void conv2_tile(
    int tile, int nEdges, int m, int half,
    const float4* __restrict__ h4, const long long* __restrict__ ei,
    const float* __restrict__ ea, const v16h (&bm)[2][2],
    const float (&sc)[8], const float (&sh)[8],
    float* __restrict__ agg2)
{
    int  e  = tile * 16 + m;
    bool ev = FULL ? true : (e < nEdges);
    int  ec = FULL ? e : (ev ? e : (nEdges - 1));

    int src = (int)__builtin_nontemporal_load(ei + ec);
    int dst = (int)__builtin_nontemporal_load(ei + nEdges + ec);
    float p0 = __builtin_nontemporal_load(ea + 2 * ec + 0);
    float p1 = __builtin_nontemporal_load(ea + 2 * ec + 1);

    float b0 = (1.0f - p1) * (1.0f - p0);
    float b1 = (1.0f - p1) * p0;
    float b2 = p1 * (1.0f - p0);
    float b3 = p1 * p0;

    // gather 8 features (two b128 loads) + fused BN1 affine
    float4 f0 = h4[src * 4 + half * 2 + 0];
    float4 f1 = h4[src * 4 + half * 2 + 1];
    float hv[8] = { f0.x, f0.y, f0.z, f0.w, f1.x, f1.y, f1.z, f1.w };
    #pragma unroll
    for (int t = 0; t < 8; ++t) {
        float v = hv[t] * sc[t] + sh[t];
        hv[t] = (FULL || ev) ? v : 0.0f;
    }

    v16h a0, a1;   // K blocks: a0 holds k=0,1 slice; a1 holds k=2,3 slice
    #pragma unroll
    for (int t = 0; t < 8; ++t) {
        a0[t]     = (_Float16)(b0 * hv[t]);
        a0[8 + t] = (_Float16)(b1 * hv[t]);
        a1[t]     = (_Float16)(b2 * hv[t]);
        a1[8 + t] = (_Float16)(b3 * hv[t]);
    }

    v8f c0, c1;
    #pragma unroll
    for (int t = 0; t < 8; ++t) { c0[t] = 0.0f; c1[t] = 0.0f; }
    c0 = __builtin_amdgcn_wmma_f32_16x16x32_f16(false, a0, false, bm[0][0], (short)0, c0, false, false);
    c0 = __builtin_amdgcn_wmma_f32_16x16x32_f16(false, a1, false, bm[1][0], (short)0, c0, false, false);
    c1 = __builtin_amdgcn_wmma_f32_16x16x32_f16(false, a0, false, bm[0][1], (short)0, c1, false, false);
    c1 = __builtin_amdgcn_wmma_f32_16x16x32_f16(false, a1, false, bm[1][1], (short)0, c1, false, false);

    int dsts[8];
    #pragma unroll
    for (int j = 0; j < 8; ++j)
        dsts[j] = __builtin_amdgcn_ds_bpermute((half * 8 + j) << 2, dst);

    #pragma unroll
    for (int j = 0; j < 8; ++j) {
        if (FULL || (tile * 16 + half * 8 + j) < nEdges) {
            float* p = agg2 + dsts[j] * 32 + m;
            atomicAdd(p,      c0[j]);
            atomicAdd(p + 16, c1[j]);
        }
    }
}

__global__ void k_conv2(const float* __restrict__ h1,
                        const long long* __restrict__ ei,
                        const float* __restrict__ ea,
                        const float* __restrict__ W2,      // [4,16,32]
                        const float* __restrict__ ss1,     // scale[16], shift[16]
                        float* __restrict__ agg2,          // [nNodes,32]
                        int nEdges)
{
    const int lane = threadIdx.x & 31;
    const int m    = lane & 15;
    const int half = lane >> 4;
    const int off  = half * 8;    // this lane's feature slice i in [off, off+8)

    // B registers: bm[kk][nn] covers K block kk (rows 32kk..32kk+31), N block nn.
    v16h bm[2][2];
    #pragma unroll
    for (int kk = 0; kk < 2; ++kk)
        #pragma unroll
        for (int nn = 0; nn < 2; ++nn)
            #pragma unroll
            for (int j = 0; j < 8; ++j) {
                int r0 = kk * 32 + half * 8 + 2 * j + (j >= 4 ? 8 : 0);
                int r1 = r0 + 1;
                int k0 = r0 >> 4, i0 = r0 & 15;
                int k1 = r1 >> 4, i1 = r1 & 15;
                bm[kk][nn][2 * j]     = (_Float16)W2[k0 * 512 + i0 * 32 + nn * 16 + m];
                bm[kk][nn][2 * j + 1] = (_Float16)W2[k1 * 512 + i1 * 32 + nn * 16 + m];
            }

    float sc[8], sh[8];
    #pragma unroll
    for (int t = 0; t < 8; ++t) { sc[t] = ss1[off + t]; sh[t] = ss1[16 + off + t]; }

    const int wave   = (int)((blockIdx.x * blockDim.x + threadIdx.x) >> 5);
    const int nWaves = (int)((gridDim.x * blockDim.x) >> 5);
    const int nTiles = (nEdges + 15) >> 4;
    const int nFull  = nEdges >> 4;
    const float4* h4 = (const float4*)h1;

    for (int tile = wave; tile < nFull; tile += nWaves)
        conv2_tile<true>(tile, nEdges, m, half, h4, ei, ea, bm, sc, sh, agg2);
    for (int tile = nFull + wave; tile < nTiles; tile += nWaves)
        conv2_tile<false>(tile, nEdges, m, half, h4, ei, ea, bm, sc, sh, agg2);
}

// ---------------------------------------------------------------- post layer2
// e = elu(agg2/max(cnt,1)); BN2 stats + per-graph raw sums (pool) + counts.
__global__ void k_post2(const float* __restrict__ agg2,
                        const float* __restrict__ cnt,
                        const long long* __restrict__ batch,
                        float* __restrict__ stats2,   // [64] = sum[32], sumsq[32]
                        float* __restrict__ gsum,     // [N_GRAPHS,32]
                        float* __restrict__ gcnt,     // [N_GRAPHS]
                        int nNodes)
{
    __shared__ float red[256];
    const int f    = threadIdx.x & 31;
    const int nl   = threadIdx.x >> 5;     // 0..7
    const int base = blockIdx.x * 32;
    if (base >= nNodes) return;

    const int lastN  = min(base + 31, nNodes - 1);
    const int gFirst = (int)batch[base];
    const int gLast  = (int)batch[lastN];
    const bool uni   = (gFirst == gLast);   // sorted batch: usually true

    float s = 0.0f, s2 = 0.0f;
    #pragma unroll
    for (int r = 0; r < 4; ++r) {
        int n = base + nl + 8 * r;
        if (n < nNodes) {
            float cv = fmaxf(cnt[n], 1.0f);
            float v  = agg2[n * 32 + f] / cv;
            float e  = v > 0.0f ? v : (expf(v) - 1.0f);
            s += e; s2 += e * e;
            if (!uni) {
                int g = (int)batch[n];
                atomicAdd(&gsum[g * 32 + f], e);
                if (f == 0) atomicAdd(&gcnt[g], 1.0f);
            }
        }
    }
    red[threadIdx.x] = s; __syncthreads();
    for (int st = 128; st >= 32; st >>= 1) {
        if ((int)threadIdx.x < st) red[threadIdx.x] += red[threadIdx.x + st];
        __syncthreads();
    }
    if (threadIdx.x < 32) {
        atomicAdd(&stats2[f], red[threadIdx.x]);
        if (uni) atomicAdd(&gsum[gFirst * 32 + f], red[threadIdx.x]);
    }
    __syncthreads();
    red[threadIdx.x] = s2; __syncthreads();
    for (int st = 128; st >= 32; st >>= 1) {
        if ((int)threadIdx.x < st) red[threadIdx.x] += red[threadIdx.x + st];
        __syncthreads();
    }
    if (threadIdx.x < 32) atomicAdd(&stats2[32 + f], red[threadIdx.x]);
    if (uni && threadIdx.x == 0) {
        atomicAdd(&gcnt[gFirst], (float)(min(32, nNodes - base)));
    }
}

// ---------------------------------------------------------------- final: BN2 + pool + FC
// BN2 is affine, pooling is a mean => normalize pooled means, then out = gm @ fc^T.
__global__ void k_out(const float* __restrict__ stats2,
                      const float* __restrict__ gsum,
                      const float* __restrict__ gcnt,
                      const float* __restrict__ g2, const float* __restrict__ b2,
                      const float* __restrict__ fc,    // [10,32]
                      float* __restrict__ out,         // [N_GRAPHS,10]
                      int nNodes)
{
    __shared__ float sc[32], sh[32];
    int t = threadIdx.x;
    if (t < 32) {
        float invn = 1.0f / (float)nNodes;
        float mu  = stats2[t] * invn;
        float var = stats2[32 + t] * invn - mu * mu;
        float inv = rsqrtf(var + EPS_BN);
        float s   = inv * g2[t];
        sc[t] = s;
        sh[t] = b2[t] - mu * s;
    }
    __syncthreads();
    if (t < N_GRAPHS * 10) {
        int g = t / 10, c = t % 10;
        float cg = fmaxf(gcnt[g], 1.0f);
        float acc = 0.0f;
        #pragma unroll
        for (int f = 0; f < 32; ++f) {
            float gm = sc[f] * (gsum[g * 32 + f] / cg) + sh[f];
            acc += gm * fc[c * 32 + f];
        }
        out[g * 10 + c] = acc;
    }
}

// ---------------------------------------------------------------- launcher
extern "C" void kernel_launch(void* const* d_in, const int* in_sizes, int n_in,
                              void* d_out, int out_size, void* d_ws, size_t ws_size,
                              hipStream_t stream)
{
    const float*     x     = (const float*)d_in[0];
    const long long* ei    = (const long long*)d_in[1];   // int64 [2,E]
    const float*     ea    = (const float*)d_in[2];
    const long long* batch = (const long long*)d_in[3];   // int64 [N]
    const float*     W1    = (const float*)d_in[4];
    const float*     g1    = (const float*)d_in[5];
    const float*     b1    = (const float*)d_in[6];
    const float*     W2    = (const float*)d_in[7];
    const float*     g2    = (const float*)d_in[8];
    const float*     b2    = (const float*)d_in[9];
    const float*     fc    = (const float*)d_in[10];

    const int nNodes = in_sizes[0] / 3;
    const int nEdges = in_sizes[1] / 2;

    float* ws = (float*)d_ws;
    size_t off = 0;
    float* agg1   = ws + off; off += (size_t)nNodes * 16;
    float* agg2   = ws + off; off += (size_t)nNodes * 32;
    float* cnt    = ws + off; off += (size_t)nNodes;
    float* stats1 = ws + off; off += 32;
    float* stats2 = ws + off; off += 64;
    float* gsum   = ws + off; off += (size_t)N_GRAPHS * 32;
    float* gcnt   = ws + off; off += N_GRAPHS;
    const int zeroN = (int)off;                  // everything accumulated-into
    float* h1     = ws + off; off += (size_t)nNodes * 16;   // 16B-aligned (off%4==0)
    float* ss1    = ws + off; off += 32;

    k_zero<<<4096, 256, 0, stream>>>(ws, zeroN);

    const int convBlocks = 2048;                 // 16384 waves over 200k tiles
    k_conv1<<<convBlocks, 256, 0, stream>>>(x, ei, ea, W1, agg1, cnt, nEdges);

    const int nodeBlocks = (nNodes + 31) / 32;
    k_post1<<<nodeBlocks, 256, 0, stream>>>(agg1, cnt, h1, stats1, nNodes);
    k_fin1<<<1, 32, 0, stream>>>(stats1, g1, b1, ss1, nNodes);

    k_conv2<<<convBlocks, 256, 0, stream>>>(h1, ei, ea, W2, ss1, agg2, nEdges);

    k_post2<<<nodeBlocks, 256, 0, stream>>>(agg2, cnt, batch, stats2, gsum, gcnt, nNodes);
    k_out<<<1, 640, 0, stream>>>(stats2, gsum, gcnt, g2, b2, fc, (float*)d_out, nNodes);
}
